// IsoMax_64149631533776
// MI455X (gfx1250) — matrix operand order repaired and checked by reading, not compiled, not run on hardware
//
#include <hip/hip_runtime.h>
#include <hip/hip_bf16.h>

typedef __bf16 bf16_t;
typedef __attribute__((ext_vector_type(8)))  __bf16 bf16x8;
typedef __attribute__((ext_vector_type(16))) __bf16 bf16x16;
typedef __attribute__((ext_vector_type(8)))  float  f32x8;

#define TILE_M 128
#define TILE_N 128
#define TILE_K 32
#define LDSW   40   // padded LDS row stride in bf16 elems (80B: keeps b128 alignment)

// ---- CDNA5 async global->LDS copy (16B per lane), tracked by ASYNCcnt ----
__device__ __forceinline__ void async_cp_b128(bf16_t* lds, const bf16_t* g)
{
    asm volatile("global_load_async_to_lds_b128 %0, %1, off"
                 :
                 : "v"((unsigned)(unsigned long long)lds), "v"(g)
                 : "memory");
}
__device__ __forceinline__ void wait_async0()
{
    asm volatile("s_wait_asynccnt 0x0" ::: "memory");
}

// ---------------- row L2-normalize -> bf16 ----------------
// One block per row, 256 threads, D must be 2048 (8 elems/thread).
__global__ __launch_bounds__(256)
void norm_rows_bf16(const float* __restrict__ in, bf16_t* __restrict__ out, int D)
{
    const int row = blockIdx.x;
    const int tid = threadIdx.x;
    const float4* ip = reinterpret_cast<const float4*>(in + (size_t)row * D);
    float4 v0 = ip[tid * 2 + 0];
    float4 v1 = ip[tid * 2 + 1];
    float ss = v0.x*v0.x + v0.y*v0.y + v0.z*v0.z + v0.w*v0.w
             + v1.x*v1.x + v1.y*v1.y + v1.z*v1.z + v1.w*v1.w;

    __shared__ float red[256];
    red[tid] = ss;
    __syncthreads();
    #pragma unroll
    for (int s = 128; s > 0; s >>= 1) {
        if (tid < s) red[tid] += red[tid + s];
        __syncthreads();
    }
    const float inv =
        __builtin_amdgcn_rcpf(fmaxf(__builtin_amdgcn_sqrtf(red[0]), 1e-12f));

    bf16x8 o;
    o[0] = (bf16_t)(v0.x * inv); o[1] = (bf16_t)(v0.y * inv);
    o[2] = (bf16_t)(v0.z * inv); o[3] = (bf16_t)(v0.w * inv);
    o[4] = (bf16_t)(v1.x * inv); o[5] = (bf16_t)(v1.y * inv);
    o[6] = (bf16_t)(v1.z * inv); o[7] = (bf16_t)(v1.w * inv);
    *reinterpret_cast<bf16x8*>(out + (size_t)row * D + tid * 8) = o;
}

// ---- per-wave 32x64 sub-tile: load fragments from LDS + 8 WMMAs ----
__device__ __forceinline__ void mma_tile(const bf16_t* As_, const bf16_t* Bs_,
                                         int wRow, int wCol, int mlane, int hi,
                                         f32x8 acc[2][4])
{
    bf16x16 afr[2], bfr[4];
    // A fragment (16x32 bf16): lanes 0-15 row m hold K{0..7,16..23},
    // lanes 16-31 hold K{8..15,24..31} -> two 16B LDS reads per frag.
    #pragma unroll
    for (int r = 0; r < 2; ++r) {
        const bf16_t* p = &As_[(wRow + r * 16 + mlane) * LDSW];
        bf16x8 c0 = *reinterpret_cast<const bf16x8*>(p + hi * 8);
        bf16x8 c1 = *reinterpret_cast<const bf16x8*>(p + 16 + hi * 8);
        #pragma unroll
        for (int i = 0; i < 8; ++i) { afr[r][i] = c0[i]; afr[r][8 + i] = c1[i]; }
    }
    // B fragment (32x16 bf16): lane n holds col n, K{0..7,16..23} (lanes 0-15)
    // or K{8..15,24..31} (lanes 16-31) -> two 16B LDS reads per frag.
    #pragma unroll
    for (int c = 0; c < 4; ++c) {
        const bf16_t* p = &Bs_[(wCol + c * 16 + mlane) * LDSW];
        bf16x8 c0 = *reinterpret_cast<const bf16x8*>(p + hi * 8);
        bf16x8 c1 = *reinterpret_cast<const bf16x8*>(p + 16 + hi * 8);
        #pragma unroll
        for (int i = 0; i < 8; ++i) { bfr[c][i] = c0[i]; bfr[c][8 + i] = c1[i]; }
    }
    #pragma unroll
    for (int r = 0; r < 2; ++r)
        #pragma unroll
        for (int c = 0; c < 4; ++c)
            acc[r][c] = __builtin_amdgcn_wmma_f32_16x16x32_bf16(
                false, afr[r], false, bfr[c], (short)0, acc[r][c], false, false);
}

// ---------------- WMMA GEMM + distance epilogue ----------------
// 256 threads = 8 wave32 in 4x2 grid; 128x128 output tile per block.
// Double-buffered LDS filled by async global->LDS (ASYNCcnt) copies.
__global__ __launch_bounds__(256)
void isomax_wmma(const bf16_t* __restrict__ X,   // [Brows][D] unit rows
                 const bf16_t* __restrict__ P,   // [Mcols][D] unit rows
                 const float*  __restrict__ scale,
                 float* __restrict__ out,        // [Brows][Mcols]
                 int Brows, int Mcols, int D)
{
    __shared__ bf16_t As[2][TILE_M * LDSW];
    __shared__ bf16_t Bs[2][TILE_N * LDSW];

    const int tid  = threadIdx.x;
    const int lane = tid & 31;
    const int wv   = tid >> 5;
    const int wRow = (wv & 3) * 32;     // 0,32,64,96
    const int wCol = (wv >> 2) * 64;    // 0,64

    const int blockRow0 = blockIdx.x * TILE_M;
    const int blockCol0 = blockIdx.y * TILE_N;

    // staging: thread t owns 32B of A and 32B of B: row = t/2, half-slab = t%2
    const int ldRow = tid >> 1;
    const int ldOff = (tid & 1) * 16;   // in bf16 elems

    int aRow = blockRow0 + ldRow; if (aRow >= Brows) aRow = Brows - 1;
    int bRow = blockCol0 + ldRow; if (bRow >= Mcols) bRow = Mcols - 1;
    const bf16_t* aG = X + (size_t)aRow * D + ldOff;
    const bf16_t* bG = P + (size_t)bRow * D + ldOff;

    bf16_t* aL0 = &As[0][ldRow * LDSW + ldOff];
    bf16_t* aL1 = &As[1][ldRow * LDSW + ldOff];
    bf16_t* bL0 = &Bs[0][ldRow * LDSW + ldOff];
    bf16_t* bL1 = &Bs[1][ldRow * LDSW + ldOff];

    const int mlane = lane & 15;
    const int hi    = lane >> 4;

    f32x8 acc[2][4] = {};

    // prologue: fill buffer 0 with K-slab 0
    async_cp_b128(aL0,     aG);
    async_cp_b128(aL0 + 8, aG + 8);
    async_cp_b128(bL0,     bG);
    async_cp_b128(bL0 + 8, bG + 8);

    for (int kt = 0; kt < D; kt += 2 * TILE_K) {
        // ---- buffer 0 holds slab kt ----
        wait_async0();           // own async writes to buf0 done
        __syncthreads();         // all waves done writing buf0 / reading buf1
        if (kt + TILE_K < D) {   // overlap: fill buf1 with slab kt+32
            const bf16_t* a = aG + kt + TILE_K;
            const bf16_t* b = bG + kt + TILE_K;
            async_cp_b128(aL1,     a);
            async_cp_b128(aL1 + 8, a + 8);
            async_cp_b128(bL1,     b);
            async_cp_b128(bL1 + 8, b + 8);
            if (kt + 2 * TILE_K < D) {   // L2 prefetch 2 slabs ahead
                __builtin_prefetch((const void*)(aG + kt + 2 * TILE_K), 0, 1);
                __builtin_prefetch((const void*)(bG + kt + 2 * TILE_K), 0, 1);
            }
        }
        mma_tile(As[0], Bs[0], wRow, wCol, mlane, hi, acc);

        // ---- buffer 1 holds slab kt+32 ----
        wait_async0();
        __syncthreads();
        if (kt + 2 * TILE_K < D) {   // fill buf0 with slab kt+64
            const bf16_t* a = aG + kt + 2 * TILE_K;
            const bf16_t* b = bG + kt + 2 * TILE_K;
            async_cp_b128(aL0,     a);
            async_cp_b128(aL0 + 8, a + 8);
            async_cp_b128(bL0,     b);
            async_cp_b128(bL0 + 8, b + 8);
        }
        mma_tile(As[1], Bs[1], wRow, wCol, mlane, hi, acc);
    }

    // epilogue: unit vectors => d2 = 2 - 2*dot; out = -|s| * sqrt(max(d2,0))
    const float s = fabsf(scale[0]);
    #pragma unroll
    for (int r = 0; r < 2; ++r) {
        #pragma unroll
        for (int c = 0; c < 4; ++c) {
            const int col = blockCol0 + wCol + c * 16 + mlane;
            if (col >= Mcols) continue;
            const int row0 = blockRow0 + wRow + r * 16 + hi * 8;  // C/D layout
            float* o = out + (size_t)row0 * Mcols + col;
            #pragma unroll
            for (int i = 0; i < 8; ++i) {
                if (row0 + i >= Brows) break;
                float d2 = fmaxf(2.0f - 2.0f * acc[r][c][i], 0.0f);
                o[(size_t)i * Mcols] = -s * __builtin_amdgcn_sqrtf(d2);
            }
        }
    }
}

extern "C" void kernel_launch(void* const* d_in, const int* in_sizes, int n_in,
                              void* d_out, int out_size, void* d_ws, size_t ws_size,
                              hipStream_t stream)
{
    const float* x  = (const float*)d_in[0];   // [B, D]
    const float* pr = (const float*)d_in[1];   // [M, D]
    const float* sc = (const float*)d_in[2];   // [1]
    const int D = 2048;
    const int B = in_sizes[0] / D;
    const int M = in_sizes[1] / D;

    bf16_t* xn = (bf16_t*)d_ws;                // B*D bf16
    bf16_t* pn = xn + (size_t)B * D;           // M*D bf16

    norm_rows_bf16<<<B, 256, 0, stream>>>(x,  xn, D);
    norm_rows_bf16<<<M, 256, 0, stream>>>(pr, pn, D);

    dim3 grid((B + TILE_M - 1) / TILE_M, (M + TILE_N - 1) / TILE_N);
    isomax_wmma<<<grid, 256, 0, stream>>>(xn, pn, sc, (float*)d_out, B, M, D);
}
